// GCL_68195490726191
// MI455X (gfx1250) — matrix operand order repaired and checked by reading, not compiled, not run on hardware
//
#include <hip/hip_runtime.h>
#include <hip/hip_bf16.h>
#include <stdint.h>

typedef _Float16 v8h  __attribute__((ext_vector_type(8)));
typedef _Float16 v16h __attribute__((ext_vector_type(16)));
typedef float    v8f  __attribute__((ext_vector_type(8)));

#define DF   64     // node feature dim
#define HIDF 128    // hidden dim
#define WPB  4      // waves per block
#define THREADS (WPB * 32)

// packed transposed weights per MLP: W1t [128n][128k] then W2t [64n][128k], f16
#define WT_ELEMS (HIDF * HIDF + DF * HIDF)   // 24576 halves = 48 KB

__device__ __forceinline__ v16h cat8(v8h lo, v8h hi8) {
    return __builtin_shufflevector(lo, hi8, 0,1,2,3,4,5,6,7,8,9,10,11,12,13,14,15);
}

__device__ __forceinline__ v8f wmma16(v16h a, v16h b, v8f c) {
    // D = A(16x32 f16) * B(32x16 f16) + C(16x16 f32)
    return __builtin_amdgcn_wmma_f32_16x16x32_f16(
        /*neg_a=*/false, a, /*neg_b=*/false, b,
        /*c_mod=*/(short)0, c, /*reuse_a=*/false, /*reuse_b=*/false);
}

// Async-copy the pre-transposed f16 weight block (48 KB) from global into LDS
// using the CDNA5 async global->LDS datapath (ASYNCcnt), then barrier.
__device__ __forceinline__ void stage_weights_async(const _Float16* __restrict__ Wt,
                                                    _Float16* sW) {
    const int CHUNKS = (WT_ELEMS * 2) / 16;              // 3072 x 16B
    // Low 32 bits of a generic pointer into the LDS aperture are the LDS byte
    // address (ISA 10.2).
    uint32_t lds_base = (uint32_t)(uintptr_t)sW;
    for (int i = threadIdx.x; i < CHUNKS; i += THREADS) {
        uint32_t lds_addr = lds_base + (uint32_t)i * 16u;
        const void* gaddr = (const char*)Wt + (size_t)i * 16u;
        asm volatile("global_load_async_to_lds_b128 %0, %1, off"
                     :: "v"(lds_addr), "v"(gaddr)
                     : "memory");
    }
    asm volatile("s_wait_asynccnt 0" ::: "memory");
    __syncthreads();
}

// ---------------------------------------------------------------------------
// prep: h (f32) -> h16 (f16); zero m_i; build transposed f16 weight blocks
// ---------------------------------------------------------------------------
__global__ void prep_kernel(const float* __restrict__ h, _Float16* __restrict__ h16,
                            float* __restrict__ m_i, long nElem) {
    long i = (long)blockIdx.x * blockDim.x + threadIdx.x;
    if (i < nElem) {
        h16[i] = (_Float16)h[i];
        m_i[i] = 0.0f;
    }
}

__global__ void prep_weights_kernel(const float* __restrict__ W1,
                                    const float* __restrict__ W2,
                                    _Float16* __restrict__ Wt) {
    int i = blockIdx.x * blockDim.x + threadIdx.x;
    if (i < HIDF * HIDF) {                       // W1 [k][n] -> Wt [n][k]
        int k = i >> 7, n = i & 127;
        Wt[n * HIDF + k] = (_Float16)W1[i];
    } else if (i < WT_ELEMS) {                   // W2 [k][n] -> Wt+off [n][k]
        int j = i - HIDF * HIDF;
        int k = j >> 6, n = j & 63;
        Wt[HIDF * HIDF + n * HIDF + k] = (_Float16)W2[j];
    }
}

// ---------------------------------------------------------------------------
// edge MLP: m_ij = relu(relu([h[r]|h[c]] @ W1 + b1) @ W2 + b2), scatter-add by r
// One wave = one tile of 16 edges.
// ---------------------------------------------------------------------------
__global__ __launch_bounds__(THREADS)
void edge_mlp_kernel(const _Float16* __restrict__ h16,
                     const int* __restrict__ eidx, int E,
                     const _Float16* __restrict__ eWt,
                     const float* __restrict__ eb1, const float* __restrict__ eb2,
                     float* __restrict__ m_i) {
    __shared__ _Float16 sW[WT_ELEMS];               // 48 KB: W1t | W2t
    __shared__ _Float16 sHid[WPB][16 * HIDF];       // 16 KB, layer-1 activations

    stage_weights_async(eWt, sW);
    const _Float16* sW1t = sW;
    const _Float16* sW2t = sW + HIDF * HIDF;

    const int wave = threadIdx.x >> 5;
    const int t    = threadIdx.x & 31;
    const int ln   = t & 15;           // A row / C-D column index
    const int hi   = t >> 4;           // half-wave selector
    const long tile = (long)blockIdx.x * WPB + wave;
    const long e0   = tile * 16;
    if (e0 >= E) return;               // wave-uniform: EXEC stays all-ones

    long eL = e0 + ln; if (eL >= E) eL = (long)E - 1;
    const int r = eidx[eL];
    const int c = eidx[(long)E + eL];

    // A fragments for layer 1: edge_input[16 x 128] = [h[r] | h[c]], K-tiles of 32
    v16h a1[4];
    {
        const _Float16* pr = h16 + (long)r * DF;
        const _Float16* pc = h16 + (long)c * DF;
#pragma unroll
        for (int kt = 0; kt < 4; ++kt) {
            const _Float16* src = (kt < 2) ? pr : pc;
            const int base = (kt & 1) * 32 + hi * 8;
            a1[kt] = cat8(*(const v8h*)(src + base), *(const v8h*)(src + base + 16));
        }
    }

    _Float16* hid = sHid[wave];

    // layer 1: hidden[16 x 128] = relu(A @ W1 + b1)
#pragma unroll
    for (int nt = 0; nt < 8; ++nt) {
        const int n = nt * 16 + ln;
        const float bias = eb1[n];
        v8f acc;
#pragma unroll
        for (int i = 0; i < 8; ++i) acc[i] = bias;
#pragma unroll
        for (int kt = 0; kt < 4; ++kt) {
            const _Float16* wp = sW1t + n * HIDF + kt * 32 + hi * 8;
            acc = wmma16(a1[kt], cat8(*(const v8h*)wp, *(const v8h*)(wp + 16)), acc);
        }
#pragma unroll
        for (int v = 0; v < 8; ++v) {
            float x = acc[v]; x = x > 0.0f ? x : 0.0f;
            hid[(v + 8 * hi) * HIDF + n] = (_Float16)x;   // D-layout -> row-major LDS
        }
    }

    // A fragments for layer 2 from LDS hidden (same-wave LDS ops are in-order)
    v16h a2[4];
#pragma unroll
    for (int kt = 0; kt < 4; ++kt) {
        const _Float16* src = hid + ln * HIDF + kt * 32 + hi * 8;
        a2[kt] = cat8(*(const v8h*)src, *(const v8h*)(src + 16));
    }

    // layer 2: m[16 x 64] = relu(hidden @ W2 + b2); scatter-add rows to m_i[r]
#pragma unroll
    for (int nt = 0; nt < 4; ++nt) {
        const int n = nt * 16 + ln;
        const float bias = eb2[n];
        v8f acc;
#pragma unroll
        for (int i = 0; i < 8; ++i) acc[i] = bias;
#pragma unroll
        for (int kt = 0; kt < 4; ++kt) {
            const _Float16* wp = sW2t + n * HIDF + kt * 32 + hi * 8;
            acc = wmma16(a2[kt], cat8(*(const v8h*)wp, *(const v8h*)(wp + 16)), acc);
        }
#pragma unroll
        for (int v = 0; v < 8; ++v) {
            const int m = v + 8 * hi;
            const long e = e0 + m;
            const int rd = __shfl(r, m, 32);     // row m's destination node
            float x = acc[v]; x = x > 0.0f ? x : 0.0f;
            if (e < E) atomicAdd(m_i + (long)rd * DF + n, x);
        }
    }
}

// ---------------------------------------------------------------------------
// node MLP: h_out = h + relu([h | m_i] @ W1 + b1) @ W2 + b2
// One wave = one tile of 16 nodes.
// ---------------------------------------------------------------------------
__global__ __launch_bounds__(THREADS)
void node_mlp_kernel(const float* __restrict__ h,
                     const _Float16* __restrict__ h16,
                     const float* __restrict__ m_i, int Nn,
                     const _Float16* __restrict__ nWt,
                     const float* __restrict__ nb1, const float* __restrict__ nb2,
                     float* __restrict__ out) {
    __shared__ _Float16 sW[WT_ELEMS];
    __shared__ _Float16 sHid[WPB][16 * HIDF];

    stage_weights_async(nWt, sW);
    const _Float16* sW1t = sW;
    const _Float16* sW2t = sW + HIDF * HIDF;

    const int wave = threadIdx.x >> 5;
    const int t    = threadIdx.x & 31;
    const int ln   = t & 15;
    const int hi   = t >> 4;
    const long tile = (long)blockIdx.x * WPB + wave;
    const long n0   = tile * 16;
    if (n0 >= Nn) return;

    long nodeL = n0 + ln; if (nodeL >= Nn) nodeL = (long)Nn - 1;

    // A fragments: [h_f16 | cvt(m_i)] for this node row
    v16h a1[4];
    {
        const _Float16* ph = h16 + nodeL * DF;
#pragma unroll
        for (int kt = 0; kt < 2; ++kt) {
            const int base = kt * 32 + hi * 8;
            a1[kt] = cat8(*(const v8h*)(ph + base), *(const v8h*)(ph + base + 16));
        }
        const float* pm = m_i + nodeL * DF;
#pragma unroll
        for (int kt = 2; kt < 4; ++kt) {
            const int base = (kt - 2) * 32 + hi * 8;
            v8h lo, hi8;
#pragma unroll
            for (int i = 0; i < 8; ++i) {
                lo[i]  = (_Float16)pm[base + i];
                hi8[i] = (_Float16)pm[base + 16 + i];
            }
            a1[kt] = cat8(lo, hi8);
        }
    }

    _Float16* hid = sHid[wave];

#pragma unroll
    for (int nt = 0; nt < 8; ++nt) {
        const int n = nt * 16 + ln;
        const float bias = nb1[n];
        v8f acc;
#pragma unroll
        for (int i = 0; i < 8; ++i) acc[i] = bias;
#pragma unroll
        for (int kt = 0; kt < 4; ++kt) {
            const _Float16* wp = sW1t + n * HIDF + kt * 32 + hi * 8;
            acc = wmma16(a1[kt], cat8(*(const v8h*)wp, *(const v8h*)(wp + 16)), acc);
        }
#pragma unroll
        for (int v = 0; v < 8; ++v) {
            float x = acc[v]; x = x > 0.0f ? x : 0.0f;
            hid[(v + 8 * hi) * HIDF + n] = (_Float16)x;
        }
    }

    v16h a2[4];
#pragma unroll
    for (int kt = 0; kt < 4; ++kt) {
        const _Float16* src = hid + ln * HIDF + kt * 32 + hi * 8;
        a2[kt] = cat8(*(const v8h*)src, *(const v8h*)(src + 16));
    }

    // layer 2 (NO final relu) + residual store
#pragma unroll
    for (int nt = 0; nt < 4; ++nt) {
        const int n = nt * 16 + ln;
        const float bias = nb2[n];
        v8f acc;
#pragma unroll
        for (int i = 0; i < 8; ++i) acc[i] = bias;
#pragma unroll
        for (int kt = 0; kt < 4; ++kt) {
            const _Float16* wp = sW2t + n * HIDF + kt * 32 + hi * 8;
            acc = wmma16(a2[kt], cat8(*(const v8h*)wp, *(const v8h*)(wp + 16)), acc);
        }
#pragma unroll
        for (int v = 0; v < 8; ++v) {
            const int m = v + 8 * hi;
            const long node = n0 + m;
            if (node < Nn)
                out[node * DF + n] = h[node * DF + n] + acc[v];
        }
    }
}

// ---------------------------------------------------------------------------
extern "C" void kernel_launch(void* const* d_in, const int* in_sizes, int n_in,
                              void* d_out, int out_size, void* d_ws, size_t ws_size,
                              hipStream_t stream) {
    const float* h    = (const float*)d_in[0];
    const int*   eidx = (const int*)d_in[1];
    const float* eW1  = (const float*)d_in[2];
    const float* eb1  = (const float*)d_in[3];
    const float* eW2  = (const float*)d_in[4];
    const float* eb2  = (const float*)d_in[5];
    const float* nW1  = (const float*)d_in[6];
    const float* nb1  = (const float*)d_in[7];
    const float* nW2  = (const float*)d_in[8];
    const float* nb2  = (const float*)d_in[9];
    float* out = (float*)d_out;

    const int Nn = in_sizes[0] / DF;
    const int E  = in_sizes[1] / 2;

    // workspace layout (256B aligned slabs):
    //   h16  : N*D f16
    //   m_i  : N*D f32
    //   eWt  : WT_ELEMS f16 (transposed edge weights)
    //   nWt  : WT_ELEMS f16 (transposed node weights)
    char* wsp = (char*)d_ws;
    _Float16* h16 = (_Float16*)wsp;
    size_t off = ((size_t)Nn * DF * sizeof(_Float16) + 255) & ~(size_t)255;
    float* m_i = (float*)(wsp + off);
    off += ((size_t)Nn * DF * sizeof(float) + 255) & ~(size_t)255;
    _Float16* eWt = (_Float16*)(wsp + off);
    off += ((size_t)WT_ELEMS * sizeof(_Float16) + 255) & ~(size_t)255;
    _Float16* nWt = (_Float16*)(wsp + off);

    const long nElem = (long)Nn * DF;
    const int  pb = 256;
    prep_kernel<<<(int)((nElem + pb - 1) / pb), pb, 0, stream>>>(h, h16, m_i, nElem);

    const int wgrid = (WT_ELEMS + pb - 1) / pb;
    prep_weights_kernel<<<wgrid, pb, 0, stream>>>(eW1, eW2, eWt);
    prep_weights_kernel<<<wgrid, pb, 0, stream>>>(nW1, nW2, nWt);

    const long etiles  = ((long)E + 15) / 16;
    const long eblocks = (etiles + WPB - 1) / WPB;
    edge_mlp_kernel<<<(int)eblocks, THREADS, 0, stream>>>(h16, eidx, E,
                                                          eWt, eb1, eb2, m_i);

    const long ntiles  = ((long)Nn + 15) / 16;
    const long nblocks = (ntiles + WPB - 1) / WPB;
    node_mlp_kernel<<<(int)nblocks, THREADS, 0, stream>>>(h, h16, m_i, Nn,
                                                          nWt, nb1, nb2, out);
}
